// SlidingChannelAttn_41394894799618
// MI455X (gfx1250) — matrix operand order repaired and checked by reading, not compiled, not run on hardware
//
#include <hip/hip_runtime.h>
#include <hip/hip_bf16.h>

// ---------------------------------------------------------------------------
// SlidingChannelAttn on MI455X (gfx1250, wave32, WMMA).
// Bandwidth-bound (~550MB moved @ 23.3TB/s ~ 24us vs few-us of f16 WMMA math):
//  - f16 intermediates, f32 WMMA accumulation
//  - all channel contractions via v_wmma_f32_16x16x32_f16
//  - fused [Q;K]x[K Q]^T Gram trick: S=QK^T, ||q||^2, ||k||^2 in one pass
//  - GEMMs: weights (A-fragments) hoisted, 4 position tiles per wave with
//    base-pointer + immediate-offset addressing so loads/stores fold into the
//    24-bit IOFFSET field (no per-element 64-bit address VALU).
// ---------------------------------------------------------------------------

typedef _Float16 half_t;
typedef __attribute__((ext_vector_type(16))) _Float16 v16h;
typedef __attribute__((ext_vector_type(8)))  float    v8f;

#define BATCH 2
#define CDIM  64
#define TDIM  8
#define SDIM  16384            // H*W = 128*128
#define FR    16               // BATCH*TDIM frames
#define HEADS 8
#define CPH   8
#define NPOS  (FR * SDIM)      // 262144 positions
#define CH_STRIDE ((long)TDIM * SDIM)   // x channel stride in elements (131072)

static __device__ __forceinline__ v8f wmma_f16(v16h a, v16h b, v8f c) {
  // 8-arg form: (neg_a, A, neg_b, B, c_mod, C, reuse_a, reuse_b)
  return __builtin_amdgcn_wmma_f32_16x16x32_f16(false, a, false, b, (short)0, c,
                                                false, false);
}

// K-offset of element e of the 16-bit A fragment, with the lane-half (hi<<4)
// term folded into the base pointer (contiguous order k = 16*hi + e).
static __device__ __forceinline__ constexpr int a_kconst(int e) {
  return e;
}

// ---------------------------------------------------------------------------
// K1: q = 1x1x1 conv (64x64 GEMM over positions), bias fused, f32 -> f16.
// grid = NPOS/64, block = 128 (4 waves; wave w owns co-tile w).
// ---------------------------------------------------------------------------
__global__ void k_q_pw(const float* __restrict__ x, const float* __restrict__ qw,
                       const float* __restrict__ qb, half_t* __restrict__ q1) {
  const int lane = threadIdx.x & 31;
  const int wave = threadIdx.x >> 5;
  const int hi = lane >> 4;
  const int ln = lane & 15;                 // A row m / B,C column n
  const long pos0 = (long)blockIdx.x << 6;  // 64 positions per block
  const int frame = (int)(pos0 >> 14);
  const int s0 = (int)(pos0 & (SDIM - 1));
  const int b_i = frame >> 3, t_i = frame & 7;
  const int co0 = wave << 4;

  // Weight A-fragments: base pointer + constant offsets (2 contiguous runs).
  const float* qwp = qw + (co0 + ln) * 64 + (hi << 4);
  v16h af[2];
#pragma unroll
  for (int kb2 = 0; kb2 < 2; ++kb2)
#pragma unroll
    for (int e = 0; e < 16; ++e)
      af[kb2][e] = (half_t)qwp[(kb2 << 5) + a_kconst(e)];
  float bias[8];
#pragma unroll
  for (int r = 0; r < 8; ++r) bias[r] = qb[co0 + r + (hi << 3)];

  const float* xp =
      x + (((long)(b_i * CDIM + (hi << 4)) * TDIM + t_i) << 14) + s0 + ln;
  v8f acc[4] = {};
#pragma unroll
  for (int kb2 = 0; kb2 < 2; ++kb2) {
    const float* xpk = xp + (long)(kb2 << 5) * CH_STRIDE;
#pragma unroll
    for (int pt = 0; pt < 4; ++pt) {
      v16h bf;
#pragma unroll
      for (int e = 0; e < 16; ++e)
        bf[e] = (half_t)xpk[(long)e * CH_STRIDE + (pt << 4)];
      acc[pt] = wmma_f16(af[kb2], bf, acc[pt]);
    }
  }
  half_t* qp = q1 + (((long)frame * CDIM + co0 + (hi << 3)) << 14) + s0 + ln;
#pragma unroll
  for (int pt = 0; pt < 4; ++pt)
#pragma unroll
    for (int r = 0; r < 8; ++r)
      qp[(long)r * SDIM + (pt << 4)] = (half_t)(acc[pt][r] + bias[r]);
}

// ---------------------------------------------------------------------------
// K3: kv = temporal conv (3,1,1), pad 1: GEMM with K = 64ci x 3 taps.
// grid = NPOS/64, block = 256 (8 waves; 128 output channels).
// Tap-boundary branch is wave-uniform -> EXEC stays all-1s for WMMA.
// ---------------------------------------------------------------------------
__global__ void k_kv_temporal(const float* __restrict__ x, const float* __restrict__ kvw,
                              const float* __restrict__ kvb, half_t* __restrict__ kv1) {
  const int lane = threadIdx.x & 31;
  const int wave = threadIdx.x >> 5;
  const int hi = lane >> 4;
  const int ln = lane & 15;
  const long pos0 = (long)blockIdx.x << 6;
  const int frame = (int)(pos0 >> 14);
  const int s0 = (int)(pos0 & (SDIM - 1));
  const int b_i = frame >> 3, t_i = frame & 7;
  const int co0 = wave << 4;

  const float* kvwp = kvw + ((co0 + ln) * 64 + (hi << 4)) * 3;
  v16h af[3][2];
#pragma unroll
  for (int tap = 0; tap < 3; ++tap)
#pragma unroll
    for (int kb2 = 0; kb2 < 2; ++kb2)
#pragma unroll
      for (int e = 0; e < 16; ++e)
        af[tap][kb2][e] = (half_t)kvwp[((kb2 << 5) + a_kconst(e)) * 3 + tap];
  float bias[8];
#pragma unroll
  for (int r = 0; r < 8; ++r) bias[r] = kvb[co0 + r + (hi << 3)];

  v8f acc[4] = {};
  for (int tap = 0; tap < 3; ++tap) {
    const int ts = t_i + tap - 1;           // uniform across wave
    if ((unsigned)ts > 7u) continue;        // zero-pad tap
    const float* xp =
        x + (((long)(b_i * CDIM + (hi << 4)) * TDIM + ts) << 14) + s0 + ln;
#pragma unroll
    for (int kb2 = 0; kb2 < 2; ++kb2) {
      const float* xpk = xp + (long)(kb2 << 5) * CH_STRIDE;
#pragma unroll
      for (int pt = 0; pt < 4; ++pt) {
        v16h bf;
#pragma unroll
        for (int e = 0; e < 16; ++e)
          bf[e] = (half_t)xpk[(long)e * CH_STRIDE + (pt << 4)];
        acc[pt] = wmma_f16(af[tap][kb2], bf, acc[pt]);
      }
    }
  }
  half_t* kvp = kv1 + (((long)frame * 128 + co0 + (hi << 3)) << 14) + s0 + ln;
#pragma unroll
  for (int pt = 0; pt < 4; ++pt)
#pragma unroll
    for (int r = 0; r < 8; ++r)
      kvp[(long)r * SDIM + (pt << 4)] = (half_t)(acc[pt][r] + bias[r]);
}

// ---------------------------------------------------------------------------
// K2/K4: depthwise 3x3 spatial conv, pad 1, bias fused. f16 in/out, f32 math.
// ---------------------------------------------------------------------------
__global__ void k_dw3x3(const half_t* __restrict__ in, const float* __restrict__ w,
                        const float* __restrict__ bias, half_t* __restrict__ out,
                        int C) {
  const long idx = (long)blockIdx.x * blockDim.x + threadIdx.x; // FR*C*SDIM
  const int s = (int)(idx & (SDIM - 1));
  const long cf = idx >> 14;
  const int c = (int)(cf % C);
  const int y = s >> 7, x0 = s & 127;
  const long base = cf << 14;
  __builtin_prefetch(&in[base + (long)SDIM], 0, 0); // global_prefetch_b8: next plane
  float acc = bias[c];
#pragma unroll
  for (int ky = 0; ky < 3; ++ky) {
    const int yy = y + ky - 1;
    if ((unsigned)yy > 127u) continue;
#pragma unroll
    for (int kx = 0; kx < 3; ++kx) {
      const int xx = x0 + kx - 1;
      if ((unsigned)xx > 127u) continue;
      acc += w[c * 9 + ky * 3 + kx] * (float)in[base + yy * 128 + xx];
    }
  }
  out[idx] = (half_t)acc;
}

// ---------------------------------------------------------------------------
// K5: fused Gram + l2norm + temperature + softmax.
// One block per (frame, head); 8 waves each own 2048 of s=16384.
// A = [Q;K] (16 rows), B = [K Q]^T  ->  D[0:8,0:8]=QK^T, D[c][8+c]=||q_c||^2,
// D[8+d][d]=||k_d||^2.  Lane-half offsets folded into base pointers so both
// fragments load as contiguous runs (b128-able). Cross-wave reduce via ds_add.
// ---------------------------------------------------------------------------
__global__ void k_attn_stats(const half_t* __restrict__ qd, const half_t* __restrict__ kvd,
                             const float* __restrict__ temp, float* __restrict__ attn) {
  const int head = blockIdx.x & 7;
  const int frame = blockIdx.x >> 3;
  const int lane = threadIdx.x & 31;
  const int wave = threadIdx.x >> 5;     // 8 waves
  const int hi = lane >> 4;
  const int ln = lane & 15;
  const int sbase0 = wave << 11;

  const half_t* ap = ((ln < 8)
      ? qd  + (((long)frame * CDIM + head * CPH + ln) << 14)
      : kvd + (((long)frame * 128 + head * CPH + (ln - 8)) << 14))
      + sbase0 + (hi << 4);
  const half_t* bp = ((ln < 8)
      ? kvd + (((long)frame * 128 + head * CPH + ln) << 14)
      : qd  + (((long)frame * CDIM + head * CPH + (ln - 8)) << 14))
      + sbase0 + (hi << 4);

  v8f acc = {};
#pragma unroll 4
  for (int i = 0; i < 64; ++i) {
    const int sb = i << 5;
    v16h af, bf;
#pragma unroll
    for (int e = 0; e < 16; ++e) af[e] = ap[sb + a_kconst(e)];
#pragma unroll
    for (int e = 0; e < 16; ++e) bf[e] = bp[sb + e];
    acc = wmma_f16(af, bf, acc);
  }

  __shared__ float red[256];             // D[16][16] cross-wave accumulator
  red[threadIdx.x] = 0.0f;
  __syncthreads();
#pragma unroll
  for (int r = 0; r < 8; ++r)
    atomicAdd(&red[((r + (hi << 3)) << 4) + ln], acc[r]);  // ds_add_f32
  __syncthreads();

  if (threadIdx.x < 8) {                 // one thread per q-row c
    const int c = threadIdx.x;
    const float tv = temp[head];
    const float qn = fmaxf(sqrtf(fmaxf(red[(c << 4) + 8 + c], 0.f)), 1e-12f);
    float row[8];
    float mx = -3.4e38f;
#pragma unroll
    for (int d = 0; d < 8; ++d) {
      const float kn = fmaxf(sqrtf(fmaxf(red[((8 + d) << 4) + d], 0.f)), 1e-12f);
      row[d] = red[(c << 4) + d] / (qn * kn) * tv;
      mx = fmaxf(mx, row[d]);
    }
    float sum = 0.f;
#pragma unroll
    for (int d = 0; d < 8; ++d) { row[d] = __expf(row[d] - mx); sum += row[d]; }
    const float inv = 1.0f / sum;
#pragma unroll
    for (int d = 0; d < 8; ++d)
      attn[((long)blockIdx.x * 8 + c) * 8 + d] = row[d] * inv;
  }
}

// ---------------------------------------------------------------------------
// K6: out = attn @ v. M=8 GEMM -> matrix units ~3% utilized; bandwidth-optimal
// VALU version with v-row reuse x8 and 16B-vectorized f16 loads/stores.
// ---------------------------------------------------------------------------
__global__ void k_apply_attn(const half_t* __restrict__ kvd, const float* __restrict__ attn,
                             half_t* __restrict__ ao) {
  const int strip = blockIdx.x & 7;
  const int fh = blockIdx.x >> 3;
  const int head = fh & 7, frame = fh >> 3;
  __shared__ float a_s[64];
  if (threadIdx.x < 64) a_s[threadIdx.x] = attn[fh * 64 + threadIdx.x];
  __syncthreads();
  const int s0 = (strip << 11) + (threadIdx.x << 3);
  const half_t* vbase =
      kvd + (((long)frame * 128 + 64 + head * CPH) << 14) + s0;

  float acc[8][8];
#pragma unroll
  for (int c = 0; c < 8; ++c)
#pragma unroll
    for (int j = 0; j < 8; ++j) acc[c][j] = 0.f;

#pragma unroll
  for (int d = 0; d < 8; ++d) {
    float vv[8];
#pragma unroll
    for (int j = 0; j < 8; ++j) vv[j] = (float)vbase[(long)d * SDIM + j];
#pragma unroll
    for (int c = 0; c < 8; ++c) {
      const float a = a_s[c * 8 + d];
#pragma unroll
      for (int j = 0; j < 8; ++j) acc[c][j] += a * vv[j];
    }
  }
  half_t* obase = ao + (((long)frame * CDIM + head * CPH) << 14) + s0;
#pragma unroll
  for (int c = 0; c < 8; ++c)
#pragma unroll
    for (int j = 0; j < 8; ++j) obase[(long)c * SDIM + j] = (half_t)acc[c][j];
}

// ---------------------------------------------------------------------------
// K7: proj = temporal conv (3,1,1) GEMM, f16 in -> f32 out (final), bias fused.
// grid = NPOS/64, block = 128 (4 waves).
// ---------------------------------------------------------------------------
__global__ void k_proj(const half_t* __restrict__ ao, const float* __restrict__ pw,
                       const float* __restrict__ pb, float* __restrict__ out) {
  const int lane = threadIdx.x & 31;
  const int wave = threadIdx.x >> 5;
  const int hi = lane >> 4;
  const int ln = lane & 15;
  const long pos0 = (long)blockIdx.x << 6;
  const int frame = (int)(pos0 >> 14);
  const int s0 = (int)(pos0 & (SDIM - 1));
  const int b_i = frame >> 3, t_i = frame & 7;
  const int co0 = wave << 4;

  const float* pwp = pw + ((co0 + ln) * 64 + (hi << 4)) * 3;
  v16h af[3][2];
#pragma unroll
  for (int tap = 0; tap < 3; ++tap)
#pragma unroll
    for (int kb2 = 0; kb2 < 2; ++kb2)
#pragma unroll
      for (int e = 0; e < 16; ++e)
        af[tap][kb2][e] = (half_t)pwp[((kb2 << 5) + a_kconst(e)) * 3 + tap];
  float bias[8];
#pragma unroll
  for (int r = 0; r < 8; ++r) bias[r] = pb[co0 + r + (hi << 3)];

  v8f acc[4] = {};
  for (int tap = 0; tap < 3; ++tap) {
    const int ts = t_i + tap - 1;           // uniform across wave
    if ((unsigned)ts > 7u) continue;
    const half_t* aop =
        ao + (((long)(b_i * TDIM + ts) * CDIM + (hi << 4)) << 14) + s0 + ln;
#pragma unroll
    for (int kb2 = 0; kb2 < 2; ++kb2) {
      const half_t* aopk = aop + ((long)(kb2 << 5) << 14);
#pragma unroll
      for (int pt = 0; pt < 4; ++pt) {
        v16h bf;
#pragma unroll
        for (int e = 0; e < 16; ++e)
          bf[e] = aopk[((long)e << 14) + (pt << 4)];
        acc[pt] = wmma_f16(af[tap][kb2], bf, acc[pt]);
      }
    }
  }
  float* outp =
      out + (((long)(b_i * CDIM + co0 + (hi << 3)) * TDIM + t_i) << 14) + s0 + ln;
#pragma unroll
  for (int pt = 0; pt < 4; ++pt)
#pragma unroll
    for (int r = 0; r < 8; ++r)
      outp[(long)r * CH_STRIDE + (pt << 4)] = acc[pt][r] + bias[r];
}

// ---------------------------------------------------------------------------
extern "C" void kernel_launch(void* const* d_in, const int* in_sizes, int n_in,
                              void* d_out, int out_size, void* d_ws, size_t ws_size,
                              hipStream_t stream) {
  const float* x       = (const float*)d_in[0];
  const float* q_w     = (const float*)d_in[1];
  const float* q_b     = (const float*)d_in[2];
  const float* q_dw_w  = (const float*)d_in[3];
  const float* q_dw_b  = (const float*)d_in[4];
  const float* kv_w    = (const float*)d_in[5];
  const float* kv_b    = (const float*)d_in[6];
  const float* kv_dw_w = (const float*)d_in[7];
  const float* kv_dw_b = (const float*)d_in[8];
  const float* proj_w  = (const float*)d_in[9];
  const float* proj_b  = (const float*)d_in[10];
  const float* temp    = (const float*)d_in[11];
  float* out = (float*)d_out;

  // Workspace carve (~201 MB): ao aliases q1 (dead after its depthwise pass).
  char* ws = (char*)d_ws;
  size_t off = 0;
  auto carve = [&](size_t bytes) {
    char* p = ws + off;
    off += (bytes + 255) & ~(size_t)255;
    return p;
  };
  half_t* q1   = (half_t*)carve((size_t)FR * CDIM * SDIM * sizeof(half_t));
  half_t* qd   = (half_t*)carve((size_t)FR * CDIM * SDIM * sizeof(half_t));
  half_t* kv1  = (half_t*)carve((size_t)FR * 128  * SDIM * sizeof(half_t));
  half_t* kvd  = (half_t*)carve((size_t)FR * 128  * SDIM * sizeof(half_t));
  float*  attn = (float*)carve((size_t)FR * HEADS * 64 * sizeof(float));
  half_t* ao   = q1;  // reuse

  k_q_pw       <<<NPOS / 64,               128, 0, stream>>>(x, q_w, q_b, q1);
  k_dw3x3      <<<(FR * CDIM * SDIM) / 256, 256, 0, stream>>>(q1, q_dw_w, q_dw_b, qd, CDIM);
  k_kv_temporal<<<NPOS / 64,               256, 0, stream>>>(x, kv_w, kv_b, kv1);
  k_dw3x3      <<<(FR * 128 * SDIM) / 256,  256, 0, stream>>>(kv1, kv_dw_w, kv_dw_b, kvd, 128);
  k_attn_stats <<<FR * HEADS,              256, 0, stream>>>(qd, kvd, temp, attn);
  k_apply_attn <<<FR * HEADS * 8,          256, 0, stream>>>(kvd, attn, ao);
  k_proj       <<<NPOS / 64,               128, 0, stream>>>(ao, proj_w, proj_b, out);
}
